// SubGraph_58213986730576
// MI455X (gfx1250) — matrix operand order repaired and checked by reading, not compiled, not run on hardware
//
#include <hip/hip_runtime.h>
#include <hip/hip_bf16.h>

// ---------------------------------------------------------------------------
// Problem constants (from reference): K=8192 clusters, S=16 nodes, IN=10, H=64
// ---------------------------------------------------------------------------
#define NCLUST   8192
#define BT       256        // glp kernel: threads per block = 8 waves
#define WAVES    8
#define FBT      256        // final kernel: threads per block = 8 waves
#define FW       8

typedef __attribute__((ext_vector_type(16))) __bf16 v16bf;
typedef __attribute__((ext_vector_type(8)))  __bf16 v8bf;
typedef __attribute__((ext_vector_type(8)))  float  v8f;
typedef __attribute__((ext_vector_type(4)))  float  v4f;
typedef int i128v __attribute__((vector_size(16)));   // v4i, matches builtin sig

#define WMMA_BF16(a, b, c) \
  __builtin_amdgcn_wmma_f32_16x16x32_bf16(false, (a), false, (b), (short)0, (c), false, false)

// --- CDNA5 async global->LDS copy (ASYNCcnt) with safe fallback ------------
#if defined(__HIP_DEVICE_COMPILE__) && defined(__has_builtin)
#if __has_builtin(__builtin_amdgcn_global_load_async_to_lds_b128)
#define HAVE_ASYNC 1
#endif
#endif
#ifndef HAVE_ASYNC
#define HAVE_ASYNC 0
#endif

struct alignas(16) Chunk16 { unsigned int w[4]; };

__device__ inline void cp_g2l_b128(const void* g, void* l) {
#if HAVE_ASYNC
  __builtin_amdgcn_global_load_async_to_lds_b128(
      (__attribute__((address_space(1))) i128v*)g,
      (__attribute__((address_space(3))) i128v*)l, 0, 0);
#else
  *(Chunk16*)l = *(const Chunk16*)g;
#endif
}
__device__ inline void wait_async_copies() {
#if HAVE_ASYNC
  asm volatile("s_wait_asynccnt 0" ::: "memory");
#endif
}

struct LayerP {
  const float *W1, *b1, *g1, *be1;
  const float *W2, *b2, *g2, *be2;
  const float *Wr, *br, *gr, *ber;
};

// ---------------------------------------------------------------------------
// Fragment access (layouts per CDNA5 ISA 7.12.2, wave32)
//   A 16x32 bf16 : lane -> row m = lane&15 ; elems e<8 -> K=k0+8*hi+e,
//                  e>=8 -> K=k0+8*hi+16+(e-8)
//   B 32x16 bf16 : lane -> col n = lane&15 ; elem e -> K=k0+8*hi+e+8*(e>>3)
//   C/D 16x16 f32: lane -> col n = lane&15 ; vgpr r -> row m = r+8*hi
//
// B matrices are pre-swizzled (once per launch, in global scratch) to
// fragment-major order so each lane's 16 elements are one contiguous,
// 32B-aligned run:  wS[((ks*T + t)*32 + lane)*16 + e]
// -> a B fragment is a single v16bf LDS load (2x ds_load_b128).
// ---------------------------------------------------------------------------
__device__ inline v16bf frag_ld(const __bf16* p) {
  return *(const v16bf*)p;
}

__device__ inline v16bf a_frag_lds(const __bf16* m, int ld, int k0, int hl, int hi) {
  const __bf16* p = m + (size_t)hl * ld + k0 + 8 * hi;
  union { v16bf v; v8bf h[2]; } u;
  u.h[0] = *(const v8bf*)(p);        // K = k0+8*hi .. +7      (16B aligned)
  u.h[1] = *(const v8bf*)(p + 16);   // K = k0+8*hi+16 .. +23  (16B aligned)
  return u.v;
}

__device__ inline void add_bias4(v8f* a, const float* b, int hl) {
#pragma unroll
  for (int t = 0; t < 4; ++t) {
    float bb = b[t * 16 + hl];
#pragma unroll
    for (int r = 0; r < 8; ++r) a[t][r] += bb;
  }
}

// Row-wise LayerNorm over the 64 features of a [16,64] tile held as 4 C-tiles.
__device__ inline void layernorm4(v8f* a, const float* g, const float* b, int hl, bool relu) {
  float s[8], q[8];
#pragma unroll
  for (int r = 0; r < 8; ++r) { s[r] = 0.f; q[r] = 0.f; }
#pragma unroll
  for (int t = 0; t < 4; ++t)
#pragma unroll
    for (int r = 0; r < 8; ++r) { float v = a[t][r]; s[r] += v; q[r] += v * v; }
#pragma unroll
  for (int r = 0; r < 8; ++r) {
#pragma unroll
    for (int m = 1; m < 16; m <<= 1) {
      s[r] += __shfl_xor(s[r], m, 32);
      q[r] += __shfl_xor(q[r], m, 32);
    }
  }
  float mean[8], rstd[8];
#pragma unroll
  for (int r = 0; r < 8; ++r) {
    mean[r] = s[r] * (1.f / 64.f);
    float var = q[r] * (1.f / 64.f) - mean[r] * mean[r];
    rstd[r] = rsqrtf(var + 1e-5f);
  }
#pragma unroll
  for (int t = 0; t < 4; ++t) {
    float gg = g[t * 16 + hl], bb = b[t * 16 + hl];
#pragma unroll
    for (int r = 0; r < 8; ++r) {
      float v = (a[t][r] - mean[r]) * rstd[r] * gg + bb;
      a[t][r] = relu ? fmaxf(v, 0.f) : v;
    }
  }
}

// ---------------------------------------------------------------------------
// Kernel 0b: one-shot weight pre-swizzle into bf16 global scratch.
//   wsw: 3 layers x [ W1s(8192) | Wrs(8192) | W2s(4096) ]  (fragment-major)
//   wsb: 3 layers x 576 f32  (b1,g1,be1,b2,g2,be2,br,gr,ber)
//   wfs: Wf swizzled, 256*128 bf16
// ---------------------------------------------------------------------------
__global__ void swizzle_kernel(LayerP L0, LayerP L1, LayerP L2,
                               const float* __restrict__ Wf,
                               __bf16* __restrict__ wsw,
                               float* __restrict__ wsb,
                               __bf16* __restrict__ wfs) {
  int idx = blockIdx.x * 256 + threadIdx.x;
  if (idx < 3 * 20480) {
    int layer = idx / 20480, j = idx - layer * 20480;
    LayerP P = (layer == 0) ? L0 : ((layer == 1) ? L1 : L2);
    int cin = (layer == 0) ? 10 : 128;
    __bf16 v;
    if (j < 16384) {                       // W1s | Wrs : 16 frags (ks*4+t)
      const float* W = (j < 8192) ? P.W1 : P.Wr;
      int i = j & 8191;
      int e = i & 15, ln = (i >> 4) & 31, fg = i >> 9;
      int k = (fg >> 2) * 32 + (ln >> 4) * 8 + (e & 7) + (e >> 3) * 16;
      int n = (fg & 3) * 16 + (ln & 15);
      v = (k < cin) ? (__bf16)W[k * 64 + n] : (__bf16)0.f;
    } else {                               // W2s : 8 frags
      int i = j - 16384;
      int e = i & 15, ln = (i >> 4) & 31, fg = i >> 9;
      int k = (fg >> 2) * 32 + (ln >> 4) * 8 + (e & 7) + (e >> 3) * 16;
      int n = (fg & 3) * 16 + (ln & 15);
      v = (__bf16)P.W2[k * 64 + n];
    }
    wsw[idx] = v;
  }
  int bidx = idx - 3 * 20480;
  if (bidx >= 0 && bidx < 3 * 576) {
    int layer = bidx / 576, j = bidx - layer * 576;
    LayerP P = (layer == 0) ? L0 : ((layer == 1) ? L1 : L2);
    const float* bsrc[9] = {P.b1, P.g1, P.be1, P.b2, P.g2, P.be2, P.br, P.gr, P.ber};
    wsb[bidx] = bsrc[j >> 6][j & 63];
  }
  int fidx = idx - (3 * 20480 + 3 * 576);
  if (fidx >= 0 && fidx < 256 * 128) {     // Wf : 64 frags (ks*8+t)
    int e = fidx & 15, ln = (fidx >> 4) & 31, fg = fidx >> 9;
    int k = (fg >> 3) * 32 + (ln >> 4) * 8 + (e & 7) + (e >> 3) * 16;
    int n = (fg & 7) * 16 + (ln & 15);
    wfs[fidx] = (__bf16)Wf[k * 128 + n];
  }
}

// ---------------------------------------------------------------------------
// Kernel 1: fully fused 3-layer GLP, one cluster per wave, 8 clusters/block.
// Weights arrive via async global->LDS copy of the pre-swizzled buffer.
// Emits xs[K,256] (sum over layers of [aggr, aggr, mean(x2), aggr]).
// ---------------------------------------------------------------------------
__global__ __launch_bounds__(BT, 1) void glp_kernel(
    const float* __restrict__ x,
    const __bf16* __restrict__ wsw, const float* __restrict__ wsb,
    float* __restrict__ xs_g) {
  __shared__ __align__(32) __bf16 wsh[20480];            // W1s|Wrs|W2s
  __shared__ __align__(16) float  biassh[576];
  __shared__ __align__(32) __bf16 xcat[WAVES][16 * 128]; // per-wave layer input
  __shared__ __align__(32) __bf16 hbuf[WAVES][16 * 64];  // per-wave h1 transpose
  __shared__ float  xsld[WAVES][256];

  const __bf16* wAs = wsh;
  const __bf16* wRs = wsh + 8192;
  const __bf16* wBs = wsh + 16384;

  const int tid = threadIdx.x;
  const int wave = tid >> 5, lane = tid & 31;
  const int hl = lane & 15, hi = lane >> 4;
  const int c = blockIdx.x * WAVES + wave;

  for (int i = tid; i < WAVES * 256; i += BT) ((float*)xsld)[i] = 0.f;

  __bf16* xc  = xcat[wave];
  __bf16* hb  = hbuf[wave];
  float*  xsl = xsld[wave];
  const v8f vzero = {};

  for (int layer = 0; layer < 3; ++layer) {
    __syncthreads();  // previous layer's weight reads done

    // ---- stage pre-swizzled weights + biases via async copies ---------
    {
      const __bf16* wsrc = wsw + (size_t)layer * 20480;   // 2560 x 16B
      const float*  bsrc = wsb + (size_t)layer * 576;     //  144 x 16B
      for (int i = tid; i < 2560; i += BT)
        cp_g2l_b128(wsrc + i * 8, wsh + i * 8);
      for (int i = tid; i < 144; i += BT)
        cp_g2l_b128(bsrc + i * 4, biassh + i * 4);
      wait_async_copies();
    }
    __syncthreads();

    // ---- x@W1 and x@Wr (shared A fragments) ---------------------------
    v8f h[4], rr[4];
#pragma unroll
    for (int t = 0; t < 4; ++t) { h[t] = vzero; rr[t] = vzero; }

    if (layer == 0) {
      // one K-step, A built from global x (f32 -> bf16, zero-padded to K=32)
      union { v16bf v; __bf16 e[16]; } u;
      const float* xr = x + (size_t)(c * 16 + hl) * 10;
#pragma unroll
      for (int e = 0; e < 16; ++e) {
        int k = (e < 8 ? e : e + 8) + 8 * hi;
        u.e[e] = (k < 10) ? (__bf16)xr[k] : (__bf16)0.f;
      }
      v16bf a = u.v;
#pragma unroll
      for (int t = 0; t < 4; ++t) {
        h[t]  = WMMA_BF16(a, frag_ld(wAs + (size_t)(t * 32 + lane) * 16), h[t]);
        rr[t] = WMMA_BF16(a, frag_ld(wRs + (size_t)(t * 32 + lane) * 16), rr[t]);
      }
    } else {
#pragma unroll
      for (int ks = 0; ks < 4; ++ks) {
        v16bf a = a_frag_lds(xc, 128, ks * 32, hl, hi);
#pragma unroll
        for (int t = 0; t < 4; ++t) {
          h[t]  = WMMA_BF16(a, frag_ld(wAs + (size_t)((ks * 4 + t) * 32 + lane) * 16), h[t]);
          rr[t] = WMMA_BF16(a, frag_ld(wRs + (size_t)((ks * 4 + t) * 32 + lane) * 16), rr[t]);
        }
      }
    }
    add_bias4(h, biassh + 0 * 64, hl);
    layernorm4(h, biassh + 1 * 64, biassh + 2 * 64, hl, true);   // LN1 + ReLU
    add_bias4(rr, biassh + 6 * 64, hl);
    layernorm4(rr, biassh + 7 * 64, biassh + 8 * 64, hl, false); // LNr

    // ---- transpose h1 into A-layout via per-wave LDS tile -------------
#pragma unroll
    for (int t = 0; t < 4; ++t)
#pragma unroll
      for (int r = 0; r < 8; ++r)
        hb[(r + 8 * hi) * 64 + t * 16 + hl] = (__bf16)h[t][r];

    // ---- h1 @ W2 ------------------------------------------------------
    v8f h2[4];
#pragma unroll
    for (int t = 0; t < 4; ++t) h2[t] = vzero;
#pragma unroll
    for (int ks = 0; ks < 2; ++ks) {
      v16bf a2 = a_frag_lds(hb, 64, ks * 32, hl, hi);
#pragma unroll
      for (int t = 0; t < 4; ++t)
        h2[t] = WMMA_BF16(a2, frag_ld(wBs + (size_t)((ks * 4 + t) * 32 + lane) * 16), h2[t]);
    }
    add_bias4(h2, biassh + 3 * 64, hl);
    layernorm4(h2, biassh + 4 * 64, biassh + 5 * 64, hl, false);

    // ---- x2 = relu(h2 + r); cluster max/mean; xs accum; next input ----
#pragma unroll
    for (int t = 0; t < 4; ++t) {
      float cmax = 0.f, csum = 0.f;  // relu output >= 0
#pragma unroll
      for (int r = 0; r < 8; ++r) {
        float v = fmaxf(h2[t][r] + rr[t][r], 0.f);
        h2[t][r] = v;
        cmax = fmaxf(cmax, v);
        csum += v;
      }
      cmax = fmaxf(cmax, __shfl_xor(cmax, 16, 32));   // combine rows 0-7 / 8-15
      csum += __shfl_xor(csum, 16, 32);
      float cmean = csum * (1.f / 16.f);
#pragma unroll
      for (int r = 0; r < 8; ++r) {
        xc[(r + 8 * hi) * 128 + t * 16 + hl]      = (__bf16)h2[t][r]; // x2
        xc[(r + 8 * hi) * 128 + 64 + t * 16 + hl] = (__bf16)cmax;     // aggr bcast
      }
      if (hi == 0) {
        int f = t * 16 + hl;
        xsl[f]        += cmax;   // mx[:64]  = colmax(x2) = aggr
        xsl[64 + f]   += cmax;   // mx[64:]  = colmax(aggr) = aggr
        xsl[128 + f]  += cmean;  // mn[:64]  = colmean(x2)
        xsl[192 + f]  += cmax;   // mn[64:]  = colmean(aggr) = aggr
      }
    }
  }

  for (int i = lane; i < 256; i += 32) xs_g[(size_t)c * 256 + i] = xsl[i];
}

// ---------------------------------------------------------------------------
// Kernel 2: out_raw = relu(xs @ Wf + bf)  [K,128], plus column sum-of-squares.
// 16 rows per wave, pre-swizzled Wf async-copied to LDS, 64 WMMA per wave.
// ---------------------------------------------------------------------------
__global__ __launch_bounds__(FBT, 1) void final_kernel(
    const float* __restrict__ xs, const __bf16* __restrict__ wfs,
    const float* __restrict__ bfv, float* __restrict__ out,
    float* __restrict__ colsq) {
  __shared__ __align__(32) __bf16 wfls[256 * 128];  // fragment-major
  __shared__ __align__(16) float  bl[128];
  const int tid = threadIdx.x, wave = tid >> 5, lane = tid & 31;
  const int hl = lane & 15, hi = lane >> 4;

  for (int i = tid; i < 4096; i += FBT)                 // 64KB in 16B chunks
    cp_g2l_b128(wfs + i * 8, wfls + i * 8);
  for (int i = tid; i < 32; i += FBT)
    cp_g2l_b128(bfv + i * 4, bl + i * 4);
  wait_async_copies();
  __syncthreads();

  const int rb = (blockIdx.x * FW + wave) * 16;

  v8f acc[8];
  const v8f vzero = {};
#pragma unroll
  for (int tn = 0; tn < 8; ++tn) acc[tn] = vzero;

#pragma unroll
  for (int ks = 0; ks < 8; ++ks) {
    // A fragment from global xs (f32 -> bf16) with vectorized b128 loads
    const float* p = xs + (size_t)(rb + hl) * 256 + ks * 32 + 8 * hi;
    v4f a0 = *(const v4f*)(p);
    v4f a1 = *(const v4f*)(p + 4);
    v4f b0 = *(const v4f*)(p + 16);
    v4f b1 = *(const v4f*)(p + 20);
    union { v16bf v; __bf16 e[16]; } u;
#pragma unroll
    for (int j = 0; j < 4; ++j) {
      u.e[j]      = (__bf16)a0[j];
      u.e[4 + j]  = (__bf16)a1[j];
      u.e[8 + j]  = (__bf16)b0[j];
      u.e[12 + j] = (__bf16)b1[j];
    }
    v16bf a = u.v;
#pragma unroll
    for (int tn = 0; tn < 8; ++tn)
      acc[tn] = WMMA_BF16(a, frag_ld(wfls + (size_t)((ks * 8 + tn) * 32 + lane) * 16), acc[tn]);
  }

#pragma unroll
  for (int tn = 0; tn < 8; ++tn) {
    int f = tn * 16 + hl;
    float bb = bl[f];
    float ss = 0.f;
#pragma unroll
    for (int r = 0; r < 8; ++r) {
      float v = fmaxf(acc[tn][r] + bb, 0.f);
      out[(size_t)(rb + r + 8 * hi) * 128 + f] = v;
      ss += v * v;
    }
    ss += __shfl_xor(ss, 16, 32);
    if (hi == 0) atomicAdd(&colsq[f], ss);
  }
}

// ---------------------------------------------------------------------------
// Kernel 3: column-wise normalization out /= (||out_col|| + 1e-8)
// ---------------------------------------------------------------------------
__global__ void norm_kernel(float* __restrict__ out, const float* __restrict__ colsq) {
  int idx = blockIdx.x * 256 + threadIdx.x;
  int f = idx & 127;
  out[idx] = out[idx] / (sqrtf(colsq[f]) + 1e-8f);
}

__global__ void zero_kernel(float* p) { p[threadIdx.x] = 0.f; }

// ---------------------------------------------------------------------------
// Host launch: detect the params flattening scheme from in_sizes.
// ---------------------------------------------------------------------------
static inline void fill_insertion(LayerP& L, void* const* d, int p) {
  L.W1 = (const float*)d[p];     L.b1 = (const float*)d[p + 1];
  L.g1 = (const float*)d[p + 2]; L.be1 = (const float*)d[p + 3];
  L.W2 = (const float*)d[p + 4]; L.b2 = (const float*)d[p + 5];
  L.g2 = (const float*)d[p + 6]; L.be2 = (const float*)d[p + 7];
  L.Wr = (const float*)d[p + 8]; L.br = (const float*)d[p + 9];
  L.gr = (const float*)d[p + 10]; L.ber = (const float*)d[p + 11];
}
static inline void fill_sorted(LayerP& L, void* const* d, int p) {
  // sorted keys: W1,W2,Wr,b1,b2,be1,be2,ber,br,g1,g2,gr
  L.W1 = (const float*)d[p];     L.W2 = (const float*)d[p + 1];
  L.Wr = (const float*)d[p + 2]; L.b1 = (const float*)d[p + 3];
  L.b2 = (const float*)d[p + 4]; L.be1 = (const float*)d[p + 5];
  L.be2 = (const float*)d[p + 6]; L.ber = (const float*)d[p + 7];
  L.br = (const float*)d[p + 8]; L.g1 = (const float*)d[p + 9];
  L.g2 = (const float*)d[p + 10]; L.gr = (const float*)d[p + 11];
}

extern "C" void kernel_launch(void* const* d_in, const int* in_sizes, int n_in,
                              void* d_out, int out_size, void* d_ws, size_t ws_size,
                              hipStream_t stream) {
  const float *x, *Wf, *bfv;
  LayerP L[3];

  if (in_sizes[0] == NCLUST * 16 * 10 && in_sizes[1] == 640) {
    // insertion order: x, [layer dicts in insertion order], Wf, bf, edges, ...
    x = (const float*)d_in[0];
    for (int l = 0; l < 3; ++l) fill_insertion(L[l], d_in, 1 + 12 * l);
    Wf = (const float*)d_in[37]; bfv = (const float*)d_in[38];
  } else if (in_sizes[0] == NCLUST * 16 * 10) {
    // x first, params as jax-sorted pytree: Wf, bf, layers(sorted keys)
    x = (const float*)d_in[0];
    Wf = (const float*)d_in[1]; bfv = (const float*)d_in[2];
    for (int l = 0; l < 3; ++l) fill_sorted(L[l], d_in, 3 + 12 * l);
  } else {
    // fully sorted top-level pytree: cluster, edge_index, num_clusters,
    // params(Wf, bf, layers-sorted), x
    Wf = (const float*)d_in[3]; bfv = (const float*)d_in[4];
    for (int l = 0; l < 3; ++l) fill_sorted(L[l], d_in, 5 + 12 * l);
    x = (const float*)d_in[n_in - 1];
  }

  const int K = out_size / 128;             // 8192 clusters
  char*   wsp   = (char*)d_ws;
  float*  xs_g  = (float*)wsp;                          // [K,256] f32 : 8 MB
  float*  colsq = (float*)(wsp + (size_t)K * 256 * 4);  // [128] f32
  __bf16* wsw   = (__bf16*)(wsp + (size_t)K * 256 * 4 + 512);        // 3*20480 bf16
  float*  wsb   = (float*) (wsp + (size_t)K * 256 * 4 + 512 + 122880); // 3*576 f32
  __bf16* wfs   = (__bf16*)(wsp + (size_t)K * 256 * 4 + 512 + 122880 + 6912); // 32768 bf16
  float*  out   = (float*)d_out;

  zero_kernel<<<1, 128, 0, stream>>>(colsq);
  swizzle_kernel<<<375, 256, 0, stream>>>(L[0], L[1], L[2], Wf, wsw, wsb, wfs);
  glp_kernel<<<K / WAVES, BT, 0, stream>>>(x, wsw, wsb, xs_g);
  final_kernel<<<K / (FW * 16), FBT, 0, stream>>>(xs_g, wfs, bfv, out, colsq);
  norm_kernel<<<(K * 128) / 256, 256, 0, stream>>>(out, colsq);
}
